// _VisualExportSelfAttention_57586921504873
// MI455X (gfx1250) — matrix operand order repaired and checked by reading, not compiled, not run on hardware
//
#include <hip/hip_runtime.h>
#include <hip/hip_bf16.h>

typedef __bf16 bf16;
typedef __attribute__((ext_vector_type(16))) __bf16 v16bf;
typedef __attribute__((ext_vector_type(8)))  __bf16 v8bf;
typedef __attribute__((ext_vector_type(8)))  float  v8f;

// TDM descriptor group types (probe: (uint32x4, int32x8, int32x4, int32x4[, int32x8], i32))
typedef __attribute__((ext_vector_type(4))) unsigned int tdm_v4u;
typedef __attribute__((ext_vector_type(8))) int          tdm_v8i;
typedef __attribute__((ext_vector_type(4))) int          tdm_v4i;

#if defined(__has_builtin)
#if __has_builtin(__builtin_amdgcn_tensor_load_to_lds) && __has_builtin(__builtin_amdgcn_s_wait_tensorcnt)
#define USE_TDM 1
#endif
#endif
#ifndef USE_TDM
#define USE_TDM 0
#endif

static __device__ __forceinline__ v8bf ld8(const bf16* p) { return *(const v8bf*)p; }

static __device__ __forceinline__ v16bf mk16(v8bf lo, v8bf hi) {
  v16bf r;
#pragma unroll
  for (int i = 0; i < 8; ++i) { r[i] = lo[i]; r[i + 8] = hi[i]; }
  return r;
}

static __device__ __forceinline__ v8f vz8() {
  v8f z = {0.f, 0.f, 0.f, 0.f, 0.f, 0.f, 0.f, 0.f};
  return z;
}

static __device__ __forceinline__ v8f wmma_bf(v16bf a, v16bf b, v8f c) {
  // (neg_a, A, neg_b, B, c_mod, C, reuse_a, reuse_b)
  return __builtin_amdgcn_wmma_f32_16x16x32_bf16(false, a, false, b, (short)0, c,
                                                 false, false);
}

#if USE_TDM
// 2D tile DMA: global (row-major, elem stride0 per row, 2-byte elems) -> LDS
// (contiguous rows of tile_d0 elems). D# fields per CDNA5 ISA 8.3/8.4.
static __device__ __forceinline__ void tdm_load_2d(unsigned lds_off, const void* gp,
                                                   unsigned tensor_d0, unsigned tensor_d1,
                                                   unsigned tile_d0, unsigned tile_d1,
                                                   unsigned stride0) {
  unsigned long long ga = (unsigned long long)gp;
  tdm_v4u g0;
  g0[0] = 1u;                                   // count=1, is_restore=0, gather=0
  g0[1] = lds_off;                              // lds_addr (bytes)
  g0[2] = (unsigned)ga;                         // global_addr[31:0]
  g0[3] = (unsigned)((ga >> 32) & 0x01FFFFFFu)  // global_addr[56:32]
          | (2u << 30);                         // type=2 ("image")
  tdm_v8i g1;
  g1[0] = (int)(1u << 16);                      // workgroup_mask=0, data_size=1 (2B)
  g1[1] = (int)((tensor_d0 & 0xFFFFu) << 16);   // tensor_dim0[15:0] @ bits 63:48
  g1[2] = (int)((tensor_d0 >> 16) | ((tensor_d1 & 0xFFFFu) << 16));
  g1[3] = (int)((tensor_d1 >> 16) | (tile_d0 << 16));   // tile_dim0 @ 127:112
  g1[4] = (int)tile_d1;                         // tile_dim1 @ 143:128, tile_dim2=0
  g1[5] = (int)stride0;                         // tensor_dim0_stride[31:0]
  g1[6] = 0;                                    // stride0[47:32]=0, stride1 lo=0
  g1[7] = 0;
  tdm_v4i z4 = {0, 0, 0, 0};
#if defined(__clang_major__) && __clang_major__ >= 23
  tdm_v8i z8 = {0, 0, 0, 0, 0, 0, 0, 0};
  __builtin_amdgcn_tensor_load_to_lds(g0, g1, z4, z4, z8, 0);
#else
  __builtin_amdgcn_tensor_load_to_lds(g0, g1, z4, z4, 0);
#endif
}
#endif

static __device__ __forceinline__ unsigned lds_offset_of(const void* p) {
  return (unsigned)(unsigned long long)p;  // generic LDS ptr: offset in low 32 bits
}

// ---------------------------------------------------------------- convert
__global__ __launch_bounds__(256) void k_cvt(const float* __restrict__ s,
                                             bf16* __restrict__ d, int n) {
  int i = blockIdx.x * 256 + threadIdx.x;
  if (i < n) d[i] = (bf16)s[i];
}

// ---------------------------------------------------------------- GEMM
// C[M,N] = A[M,K] * W[N,K]^T + bias[N]
// block tile 256x128, 8 waves (4x2), wave tile 64x64 (4x4 WMMA accumulators)
// TDM double-buffered: DMA of tile k+1 overlaps WMMA on tile k.
template <bool BF16OUT>
__global__ __launch_bounds__(256) void k_gemm(const bf16* __restrict__ A,
                                              const bf16* __restrict__ W,
                                              const float* __restrict__ bias,
                                              void* __restrict__ out,
                                              int M, int N, int K) {
  constexpr int BK = 32;
  __shared__ alignas(16) bf16 sA[2][256 * BK];
  __shared__ alignas(16) bf16 sW[2][128 * BK];

  const int tid  = threadIdx.x;
  const int lane = tid & 31;
  const int wid  = tid >> 5;
  const int half = lane >> 4;
  const int l15  = lane & 15;
  const int m0   = (wid >> 1) * 64;   // wave row origin in 256-row tile
  const int n0   = (wid & 1) * 64;    // wave col origin in 128-col tile
  const int brow = blockIdx.y * 256;
  const int bcol = blockIdx.x * 128;

  v8f acc[4][4];
#pragma unroll
  for (int i = 0; i < 4; ++i)
#pragma unroll
    for (int j = 0; j < 4; ++j) acc[i][j] = vz8();

#if USE_TDM
  const unsigned ldsA0 = lds_offset_of(sA[0]);
  const unsigned ldsA1 = lds_offset_of(sA[1]);
  const unsigned ldsW0 = lds_offset_of(sW[0]);
  const unsigned ldsW1 = lds_offset_of(sW[1]);

  // prologue: fill buffer 0
  if (wid == 0) {
    tdm_load_2d(ldsA0, A + (size_t)brow * K, (unsigned)K, (unsigned)M, BK, 256,
                (unsigned)K);
    tdm_load_2d(ldsW0, W + (size_t)bcol * K, (unsigned)K, (unsigned)N, BK, 128,
                (unsigned)K);
    __builtin_amdgcn_s_wait_tensorcnt(0);
  }
  __syncthreads();

  int cur = 0;
  for (int k0 = 0; k0 < K; k0 += BK) {
    // prefetch next tile into the other buffer (all waves already passed the
    // barrier proving nobody still reads it)
    if (wid == 0 && (k0 + BK) < K) {
      tdm_load_2d(cur ? ldsA0 : ldsA1, A + (size_t)brow * K + k0 + BK,
                  (unsigned)K, (unsigned)M, BK, 256, (unsigned)K);
      tdm_load_2d(cur ? ldsW0 : ldsW1, W + (size_t)bcol * K + k0 + BK,
                  (unsigned)K, (unsigned)N, BK, 128, (unsigned)K);
    }
    const bf16* cA = sA[cur];
    const bf16* cW = sW[cur];

    v16bf af[4];
#pragma unroll
    for (int mt = 0; mt < 4; ++mt) {
      const bf16* p = cA + (m0 + mt * 16 + l15) * BK + half * 8;
      af[mt] = mk16(ld8(p), ld8(p + 16));
    }
    v16bf bfr[4];
#pragma unroll
    for (int nt = 0; nt < 4; ++nt) {
      const bf16* p = cW + (n0 + nt * 16 + l15) * BK + half * 8;
      bfr[nt] = mk16(ld8(p), ld8(p + 16));
    }
#pragma unroll
    for (int mt = 0; mt < 4; ++mt)
#pragma unroll
      for (int nt = 0; nt < 4; ++nt)
        acc[mt][nt] = wmma_bf(af[mt], bfr[nt], acc[mt][nt]);

    __syncthreads();                       // all reads of buf[cur] done
    if (wid == 0) __builtin_amdgcn_s_wait_tensorcnt(0);  // buf[cur^1] ready
    __syncthreads();
    cur ^= 1;
  }
#else
  for (int k0 = 0; k0 < K; k0 += BK) {
    __syncthreads();
    {  // cooperative fallback: A 256 rows (1 row/thread), W 128 rows (1/2 row)
      const uint4* g = (const uint4*)(A + (size_t)(brow + tid) * K + k0);
      uint4* l = (uint4*)(sA[0] + tid * BK);
      l[0] = g[0]; l[1] = g[1]; l[2] = g[2]; l[3] = g[3];
      const int wr = tid >> 1, wsg = (tid & 1) * 16;
      const uint4* gw = (const uint4*)(W + (size_t)(bcol + wr) * K + k0 + wsg);
      uint4* lw = (uint4*)(sW[0] + wr * BK + wsg);
      lw[0] = gw[0]; lw[1] = gw[1];
    }
    __syncthreads();
    v16bf af[4];
#pragma unroll
    for (int mt = 0; mt < 4; ++mt) {
      const bf16* p = sA[0] + (m0 + mt * 16 + l15) * BK + half * 8;
      af[mt] = mk16(ld8(p), ld8(p + 16));
    }
    v16bf bfr[4];
#pragma unroll
    for (int nt = 0; nt < 4; ++nt) {
      const bf16* p = sW[0] + (n0 + nt * 16 + l15) * BK + half * 8;
      bfr[nt] = mk16(ld8(p), ld8(p + 16));
    }
#pragma unroll
    for (int mt = 0; mt < 4; ++mt)
#pragma unroll
      for (int nt = 0; nt < 4; ++nt)
        acc[mt][nt] = wmma_bf(af[mt], bfr[nt], acc[mt][nt]);
  }
#endif

  // epilogue: C layout — VGPR r: row = r + 8*half, col = l15
#pragma unroll
  for (int nt = 0; nt < 4; ++nt) {
    const int colg = bcol + n0 + nt * 16 + l15;
    const float bv = bias[colg];
#pragma unroll
    for (int mt = 0; mt < 4; ++mt) {
#pragma unroll
      for (int r = 0; r < 8; ++r) {
        const int rowg = brow + m0 + mt * 16 + r + 8 * half;
        const float v = acc[mt][nt][r] + bv;
        if (BF16OUT)
          ((bf16*)out)[(size_t)rowg * N + colg] = (bf16)v;
        else
          ((float*)out)[(size_t)rowg * N + colg] = v;
      }
    }
  }
}

// ---------------------------------------------------------------- attention
// qkv: bf16 [B*S, 3E]; per block: one (b,h), 64 q rows (4 waves x 16 rows).
// Flash-style online softmax (base-2 domain) over key tiles of 32.
__global__ __launch_bounds__(128) void k_attn(const bf16* __restrict__ qkv,
                                              const unsigned char* __restrict__ mask,
                                              bf16* __restrict__ ctx) {
  constexpr int S = 1024, E = 1024, TE = 3072;
  const float SCALE2 = 0.125f * 1.44269504088896340736f;  // (1/sqrt(64))*log2(e)

  __shared__ alignas(16) bf16 sK[32 * 64];     // [key][d]
  __shared__ alignas(16) bf16 sVt[64 * 32];    // [d][key] (transposed V)
  __shared__ alignas(16) bf16 sP[4 * 16 * 32]; // per-wave P tile

  const int tid  = threadIdx.x;
  const int lane = tid & 31;
  const int wid  = tid >> 5;
  const int half = lane >> 4;
  const int l15  = lane & 15;

  const int b  = blockIdx.y >> 4;
  const int h  = blockIdx.y & 15;
  const int q0 = blockIdx.x * 64;

  // preload Q fragments (16 rows x 64 d, two K=32 slices)
  const int qrow = q0 + wid * 16 + l15;
  const bf16* qp = qkv + (size_t)(b * S + qrow) * TE + h * 64;
  v16bf qa[2];
#pragma unroll
  for (int ds = 0; ds < 2; ++ds) {
    const bf16* p = qp + ds * 32 + half * 8;
    qa[ds] = mk16(ld8(p), ld8(p + 16));
  }

  float m_run[8], l_run[8];
  v8f oacc[4];
#pragma unroll
  for (int r = 0; r < 8; ++r) { m_run[r] = -1e30f; l_run[r] = 0.f; }
#pragma unroll
  for (int dt = 0; dt < 4; ++dt) oacc[dt] = vz8();

  const int krow = tid >> 2;  // 0..31
  const int seg  = tid & 3;   // 16-elem segment of d

#if USE_TDM
  const unsigned ldsK = lds_offset_of(sK);
#endif

  for (int kt0 = 0; kt0 < S; kt0 += 32) {
    __syncthreads();
#if USE_TDM
    if (wid == 0) {  // K tile via TDM: 32 rows x 64 elems, row stride TE
      tdm_load_2d(ldsK, qkv + (size_t)(b * S + kt0) * TE + E + h * 64,
                  (unsigned)TE, (unsigned)(8 * S), 64, 32, (unsigned)TE);
    }
#else
    {  // K tile -> LDS row-major [key][d]
      const uint4* g = (const uint4*)(qkv + (size_t)(b * S + kt0 + krow) * TE +
                                      E + h * 64 + seg * 16);
      uint4* l = (uint4*)(sK + krow * 64 + seg * 16);
      l[0] = g[0]; l[1] = g[1];
    }
#endif
    {  // V tile -> LDS transposed [d][key]
      const bf16* g = qkv + (size_t)(b * S + kt0 + krow) * TE + 2 * E + h * 64 + seg * 16;
#pragma unroll
      for (int i = 0; i < 16; ++i) sVt[(seg * 16 + i) * 32 + krow] = g[i];
    }
#if USE_TDM
    if (wid == 0) __builtin_amdgcn_s_wait_tensorcnt(0);
#endif
    __syncthreads();

    // scores: Q(16x64) @ K_tile^T -> two 16x16 tiles
    v8f sc[2];
#pragma unroll
    for (int nt = 0; nt < 2; ++nt) {
      sc[nt] = vz8();
#pragma unroll
      for (int ds = 0; ds < 2; ++ds) {
        const bf16* p = sK + (nt * 16 + l15) * 64 + ds * 32 + half * 8;
        v16bf bb = mk16(ld8(p), ld8(p + 16));
        sc[nt] = wmma_bf(qa[ds], bb, sc[nt]);
      }
    }

    // scale into log2 domain + key-padding mask (per column == per lane)
    float sv[2][8];
#pragma unroll
    for (int nt = 0; nt < 2; ++nt) {
      const int key = kt0 + nt * 16 + l15;
      const bool mk = mask[(size_t)b * S + key] != 0;
#pragma unroll
      for (int r = 0; r < 8; ++r) {
        const float v = sc[nt][r] * SCALE2;
        sv[nt][r] = mk ? -1e9f : v;
      }
    }

    // online softmax: row stats live per lane for rows r + 8*half
    float fac[8];
#pragma unroll
    for (int r = 0; r < 8; ++r) {
      float v = fmaxf(sv[0][r], sv[1][r]);
      v = fmaxf(v, __shfl_xor(v, 1, 32));
      v = fmaxf(v, __shfl_xor(v, 2, 32));
      v = fmaxf(v, __shfl_xor(v, 4, 32));
      v = fmaxf(v, __shfl_xor(v, 8, 32));
      const float mnew = fmaxf(m_run[r], v);
      fac[r] = exp2f(m_run[r] - mnew);
      m_run[r] = mnew;
    }
#pragma unroll
    for (int nt = 0; nt < 2; ++nt)
#pragma unroll
      for (int r = 0; r < 8; ++r) sv[nt][r] = exp2f(sv[nt][r] - m_run[r]);
#pragma unroll
    for (int r = 0; r < 8; ++r) {
      float s = sv[0][r] + sv[1][r];
      s += __shfl_xor(s, 1, 32);
      s += __shfl_xor(s, 2, 32);
      s += __shfl_xor(s, 4, 32);
      s += __shfl_xor(s, 8, 32);
      l_run[r] = l_run[r] * fac[r] + s;
    }
#pragma unroll
    for (int dt = 0; dt < 4; ++dt)
#pragma unroll
      for (int r = 0; r < 8; ++r) oacc[dt][r] *= fac[r];

    // P (C layout) -> per-wave LDS tile -> reload as A fragment
    bf16* pw = sP + wid * 512;
#pragma unroll
    for (int nt = 0; nt < 2; ++nt)
#pragma unroll
      for (int r = 0; r < 8; ++r)
        pw[(r + 8 * half) * 32 + nt * 16 + l15] = (bf16)sv[nt][r];

    const bf16* prd = sP + wid * 512 + l15 * 32 + half * 8;
    v16bf pf = mk16(ld8(prd), ld8(prd + 16));

    // O += P(16x32) @ V_tile(32x64): Vt is [d][key], B col n = d index
#pragma unroll
    for (int dt = 0; dt < 4; ++dt) {
      const bf16* p = sVt + (dt * 16 + l15) * 32 + half * 8;
      v16bf bb = mk16(ld8(p), ld8(p + 16));
      oacc[dt] = wmma_bf(pf, bb, oacc[dt]);
    }
  }

  float inv[8];
#pragma unroll
  for (int r = 0; r < 8; ++r) inv[r] = l_run[r] > 0.f ? 1.f / l_run[r] : 0.f;
#pragma unroll
  for (int dt = 0; dt < 4; ++dt) {
#pragma unroll
    for (int r = 0; r < 8; ++r) {
      const int row = q0 + wid * 16 + r + 8 * half;
      ctx[(size_t)(b * S + row) * E + h * 64 + dt * 16 + l15] =
          (bf16)(oacc[dt][r] * inv[r]);
    }
  }
}

// ---------------------------------------------------------------- launch
extern "C" void kernel_launch(void* const* d_in, const int* in_sizes, int n_in,
                              void* d_out, int out_size, void* d_ws, size_t ws_size,
                              hipStream_t stream) {
  const float* x     = (const float*)d_in[0];
  const float* in_w  = (const float*)d_in[1];
  const float* in_b  = (const float*)d_in[2];
  const float* out_w = (const float*)d_in[3];
  const float* out_b = (const float*)d_in[4];
  const unsigned char* kpm = (const unsigned char*)d_in[5];

  constexpr int S = 1024, E = 1024, TE = 3072;
  constexpr int M = 8 * S;  // 8192

  // workspace layout (bytes)
  char* ws = (char*)d_ws;
  bf16* xb   = (bf16*)(ws);              // 8,388,608 elems
  bf16* wqb  = (bf16*)(ws + 16777216);   // 3,145,728 elems
  bf16* wob  = (bf16*)(ws + 23068672);   // 1,048,576 elems
  bf16* qkvb = (bf16*)(ws + 25165824);   // 8192*3072
  bf16* ctxb = (bf16*)(ws + 75497472);   // 8192*1024

  k_cvt<<<8388608 / 256, 256, 0, stream>>>(x, xb, 8388608);
  k_cvt<<<3145728 / 256, 256, 0, stream>>>(in_w, wqb, 3145728);
  k_cvt<<<1048576 / 256, 256, 0, stream>>>(out_w, wob, 1048576);

  dim3 g1(TE / 128, M / 256);
  k_gemm<true><<<g1, 256, 0, stream>>>(xb, wqb, in_b, (void*)qkvb, M, TE, E);

  dim3 g2(S / 64, 8 * 16);
  k_attn<<<g2, 128, 0, stream>>>(qkvb, kpm, ctxb);

  dim3 g3(E / 128, M / 256);
  k_gemm<false><<<g3, 256, 0, stream>>>(ctxb, wob, out_b, d_out, M, E, E);
}
